// Attention_22067541967525
// MI455X (gfx1250) — matrix operand order repaired and checked by reading, not compiled, not run on hardware
//
#include <hip/hip_runtime.h>

#define BATCH 8
#define SEQ   2048
#define HID   256

typedef __attribute__((ext_vector_type(16))) __bf16 v16bf;
typedef __attribute__((ext_vector_type(8)))  float  v8f;

// ---- helpers -------------------------------------------------------------

// float -> bf16 via native conversion (gfx1250 has hw bf16 cvt)
__device__ __forceinline__ __bf16 f2bf(float f) {
    return static_cast<__bf16>(f);
}

// A-matrix (16x32 bf16) element->K mapping, per ISA 7.12.2:
// VGPR0..3 : K = 2j,2j+1 (+8 for lanes 16..31) ; VGPR4..7 : K = 16+2(j-4),... (+8)
__device__ __forceinline__ int a_kofs(int e, int hf) {
    int vg = e >> 1;
    int base = (vg < 4) ? (2 * vg) : (16 + 2 * (vg - 4));
    return base + hf * 8 + (e & 1);
}
// B-matrix (32x16 bf16): same K mapping as A (e<8: K=e+8*kh; e>=8: K=16+(e-8)+8*kh)
__device__ __forceinline__ int b_kofs(int e, int kh) {
    int vg = e >> 1;
    int base = (vg < 4) ? (2 * vg) : (16 + 2 * (vg - 4));
    return base + kh * 8 + (e & 1);
}

// A fragment from f32 row-major src (row stride ld), rows m0.., cols k0..
__device__ __forceinline__ v16bf ldA_f32(const float* p, int ld, int m0, int k0, int lane) {
    int M = m0 + (lane & 15), hf = lane >> 4;
    v16bf a;
#pragma unroll
    for (int e = 0; e < 16; ++e) a[e] = f2bf(p[(size_t)M * ld + k0 + a_kofs(e, hf)]);
    return a;
}
__device__ __forceinline__ v16bf ldA_bf(const __bf16* p, int ld, int m0, int k0, int lane) {
    int M = m0 + (lane & 15), hf = lane >> 4;
    v16bf a;
#pragma unroll
    for (int e = 0; e < 16; ++e) a[e] = p[(size_t)M * ld + k0 + a_kofs(e, hf)];
    return a;
}
// B fragment with B[K][N] = src[(n0+N)*ld + k0+K]   (src row-major, transposed use)
__device__ __forceinline__ v16bf ldBT_f32(const float* p, int ld, int n0, int k0, int lane) {
    int N = n0 + (lane & 15), kh = lane >> 4;
    v16bf b;
#pragma unroll
    for (int e = 0; e < 16; ++e) b[e] = f2bf(p[(size_t)N * ld + k0 + b_kofs(e, kh)]);
    return b;
}
__device__ __forceinline__ v16bf ldBT_bf(const __bf16* p, int ld, int n0, int k0, int lane) {
    int N = n0 + (lane & 15), kh = lane >> 4;
    v16bf b;
#pragma unroll
    for (int e = 0; e < 16; ++e) b[e] = p[(size_t)N * ld + k0 + b_kofs(e, kh)];
    return b;
}
// B fragment with B[K][N] = src[(k0+K)*ld + n0+N]   (src row-major, direct use)
__device__ __forceinline__ v16bf ldB_bf(const __bf16* p, int ld, int k0, int n0, int lane) {
    int N = n0 + (lane & 15), kh = lane >> 4;
    v16bf b;
#pragma unroll
    for (int e = 0; e < 16; ++e) b[e] = p[(size_t)(k0 + b_kofs(e, kh)) * ld + N];
    return b;
}

#define WMMA_BF16(a, b, c) \
    __builtin_amdgcn_wmma_f32_16x16x32_bf16(false, (a), false, (b), (short)0, (c), false, false)

// ---- kernel 1: q/k/v = x @ W^T + b, stored as bf16 -----------------------

__global__ __launch_bounds__(32)
void qkv_kernel(const float* __restrict__ x,
                const float* __restrict__ Wq, const float* __restrict__ bq,
                const float* __restrict__ Wk, const float* __restrict__ bk,
                const float* __restrict__ Wv, const float* __restrict__ bv,
                __bf16* __restrict__ qb, __bf16* __restrict__ kb, __bf16* __restrict__ vb) {
    int lane = threadIdx.x & 31;
    int m0 = blockIdx.x * 16;          // row in flattened (B*S, H)
    int n0 = blockIdx.y * 16;          // output feature
    v8f cq = {}, ck = {}, cv = {};
#pragma unroll
    for (int k0 = 0; k0 < HID; k0 += 32) {
        v16bf a   = ldA_f32(x, HID, m0, k0, lane);
        v16bf fbq = ldBT_f32(Wq, HID, n0, k0, lane);
        v16bf fbk = ldBT_f32(Wk, HID, n0, k0, lane);
        v16bf fbv = ldBT_f32(Wv, HID, n0, k0, lane);
        cq = WMMA_BF16(a, fbq, cq);
        ck = WMMA_BF16(a, fbk, ck);
        cv = WMMA_BF16(a, fbv, cv);
    }
    int n = n0 + (lane & 15);
    int hf = lane >> 4;
    float biq = bq[n], bik = bk[n], biv = bv[n];
#pragma unroll
    for (int r = 0; r < 8; ++r) {
        int row = m0 + r + hf * 8;
        size_t idx = (size_t)row * HID + n;
        qb[idx] = f2bf(cq[r] + biq);
        kb[idx] = f2bf(ck[r] + bik);
        vb[idx] = f2bf(cv[r] + biv);
    }
}

// ---- kernel 2: per-key-column max & sum(exp) over the query axis ---------

__global__ __launch_bounds__(32)
void stats_kernel(const __bf16* __restrict__ qb, const __bf16* __restrict__ kb,
                  float* __restrict__ colmax, float* __restrict__ colsum) {
    int lane = threadIdx.x & 31;
    int b  = blockIdx.y;
    int n0 = blockIdx.x * 16;          // key column tile
    const __bf16* qbb = qb + (size_t)b * SEQ * HID;
    const __bf16* kbb = kb + (size_t)b * SEQ * HID;

    v16bf bf[8];                       // hoisted k-matrix fragments (fixed per wave)
#pragma unroll
    for (int t = 0; t < 8; ++t) bf[t] = ldBT_bf(kbb, HID, n0, t * 32, lane);

    const float scale = 1.0f / (float)HID;
    float m = -3.0e38f, ssum = 0.0f;
    for (int q0 = 0; q0 < SEQ; q0 += 16) {
        // prefetch next q tile (emits global_prefetch_b8; overlaps the wmma chain)
        if (q0 + 16 < SEQ)
            __builtin_prefetch(qbb + (size_t)(q0 + 16 + (lane & 15)) * HID, 0, 1);
        v8f c = {};
#pragma unroll
        for (int t = 0; t < 8; ++t) {
            v16bf a = ldA_bf(qbb, HID, q0, t * 32, lane);
            c = WMMA_BF16(a, bf[t], c);
        }
        float tmax = -3.0e38f;
#pragma unroll
        for (int r = 0; r < 8; ++r) { c[r] *= scale; tmax = fmaxf(tmax, c[r]); }
        tmax = fmaxf(tmax, __shfl_xor(tmax, 16, 32));   // combine wave halves (same column)
        float nm = fmaxf(m, tmax);
        float pe = 0.0f;
#pragma unroll
        for (int r = 0; r < 8; ++r) pe += __expf(c[r] - nm);
        pe += __shfl_xor(pe, 16, 32);
        ssum = ssum * __expf(m - nm) + pe;
        m = nm;
    }
    if (lane < 16) {
        colmax[(size_t)b * SEQ + n0 + lane] = m;
        colsum[(size_t)b * SEQ + n0 + lane] = ssum;
    }
}

// ---- kernel 3: out = colnorm(exp(scores)) @ v ----------------------------

#define STAGE_LDW 40   // bf16 row stride: 80B, 16B-aligned, conflict-free (20*M mod 64 distinct)

__global__ __launch_bounds__(32)
void out_kernel(const __bf16* __restrict__ qb, const __bf16* __restrict__ kb,
                const __bf16* __restrict__ vb,
                const float* __restrict__ colmax, const float* __restrict__ colsum,
                float* __restrict__ out) {
    __shared__ __bf16 stage[16 * STAGE_LDW];  // 16 q-rows x 32 k-cols (padded)
    int lane = threadIdx.x & 31;
    int b  = blockIdx.z;
    int q0 = blockIdx.x * 16;
    int h0 = blockIdx.y * 16;
    const __bf16* qbb = qb + (size_t)b * SEQ * HID;
    const __bf16* kbb = kb + (size_t)b * SEQ * HID;
    const __bf16* vbb = vb + (size_t)b * SEQ * HID;
    const float*  cm  = colmax + (size_t)b * SEQ;
    const float*  cs  = colsum + (size_t)b * SEQ;

    v16bf aq[8];                       // hoisted q fragments (fixed per wave)
#pragma unroll
    for (int t = 0; t < 8; ++t) aq[t] = ldA_bf(qbb, HID, q0, t * 32, lane);

    v8f acc = {};
    int hf = lane >> 4, nl = lane & 15;
    const float scale = 1.0f / (float)HID;

    for (int kbase = 0; kbase < SEQ; kbase += 32) {
#pragma unroll
        for (int kt = 0; kt < 2; ++kt) {
            int kcol0 = kbase + kt * 16;
            v8f c = {};
#pragma unroll
            for (int t = 0; t < 8; ++t) {
                v16bf fk = ldBT_bf(kbb, HID, kcol0, t * 32, lane);
                c = WMMA_BF16(aq[t], fk, c);
            }
            float mcol = cm[kcol0 + nl];
            float inv  = 1.0f / cs[kcol0 + nl];
#pragma unroll
            for (int r = 0; r < 8; ++r) {
                float w = __expf(c[r] * scale - mcol) * inv;
                stage[(r + hf * 8) * STAGE_LDW + kt * 16 + nl] = f2bf(w);
            }
        }
        __syncthreads();
        v16bf aw;
        {
            int M = lane & 15;
#pragma unroll
            for (int e = 0; e < 16; ++e) aw[e] = stage[M * STAGE_LDW + a_kofs(e, hf)];
        }
        v16bf fv = ldB_bf(vbb, HID, kbase, h0, lane);
        acc = WMMA_BF16(aw, fv, acc);
        __syncthreads();
    }
#pragma unroll
    for (int r = 0; r < 8; ++r) {
        int row = q0 + r + hf * 8;
        out[((size_t)b * SEQ + row) * HID + h0 + nl] = acc[r];
    }
}

// ---- launcher ------------------------------------------------------------

extern "C" void kernel_launch(void* const* d_in, const int* in_sizes, int n_in,
                              void* d_out, int out_size, void* d_ws, size_t ws_size,
                              hipStream_t stream) {
    const float* x  = (const float*)d_in[0];
    const float* Wq = (const float*)d_in[1];
    const float* bq = (const float*)d_in[2];
    const float* Wk = (const float*)d_in[3];
    const float* bk = (const float*)d_in[4];
    const float* Wv = (const float*)d_in[5];
    const float* bv = (const float*)d_in[6];
    float* out = (float*)d_out;

    const size_t NTOK = (size_t)BATCH * SEQ;          // 16384
    __bf16* qb = (__bf16*)d_ws;                       // NTOK*HID bf16 = 8 MB
    __bf16* kb = qb + NTOK * HID;                     // 8 MB
    __bf16* vb = kb + NTOK * HID;                     // 8 MB
    float* colmax = (float*)(vb + NTOK * HID);        // NTOK f32
    float* colsum = colmax + NTOK;                    // NTOK f32  (~25.3 MB total)

    dim3 g1(NTOK / 16, HID / 16);                     // 1024 x 16 tiles
    qkv_kernel<<<g1, 32, 0, stream>>>(x, Wq, bq, Wk, bk, Wv, bv, qb, kb, vb);

    dim3 g2(SEQ / 16, BATCH);                         // 128 x 8 column tiles
    stats_kernel<<<g2, 32, 0, stream>>>(qb, kb, colmax, colsum);

    dim3 g3(SEQ / 16, HID / 16, BATCH);               // 128 x 16 x 8 output tiles
    out_kernel<<<g3, 32, 0, stream>>>(qb, kb, vb, colmax, colsum, out);
}
